// InfoNCE_31774168056172
// MI455X (gfx1250) — matrix operand order repaired and checked by reading, not compiled, not run on hardware
//
#include <hip/hip_runtime.h>
#include <hip/hip_bf16.h>
#include <math.h>

// ---- types for WMMA (CDNA5, wave32) ----
typedef __attribute__((ext_vector_type(16))) __bf16 v16bf;
typedef __attribute__((ext_vector_type(8)))  __bf16 v8bf;
typedef __attribute__((ext_vector_type(8)))  float  v8f;

#define DIM        256
#define INV_T      10.0f
#define EPSN       1e-12f
#define COL_CHUNKS 16

// float -> bf16 bits, round-to-nearest-even
static __device__ __forceinline__ unsigned short f2bf_bits(float f) {
    unsigned u = __float_as_uint(f);
    unsigned r = u + 0x7FFFu + ((u >> 16) & 1u);
    return (unsigned short)(r >> 16);
}

// ------------------------------------------------------------------
// Kernel 1: per pair-row i (one wave each): L2-normalize z1[i], z2[i],
// emit bf16 rows of Z = [z1n ; z2n], and exact f32 dot(z1n[i], z2n[i]).
// ------------------------------------------------------------------
__global__ void infonce_normalize(const float* __restrict__ z1,
                                  const float* __restrict__ z2,
                                  unsigned short* __restrict__ zbf,
                                  float* __restrict__ dpos, int B) {
    const int wave = (blockIdx.x * blockDim.x + threadIdx.x) >> 5;
    const int lane = threadIdx.x & 31;
    if (wave >= B) return;

    const float* r1 = z1 + (size_t)wave * DIM;
    const float* r2 = z2 + (size_t)wave * DIM;
    const int base = lane * 8;   // 8 elements per lane, 32 lanes * 8 = 256

    float x1[8], x2[8];
    float n1 = 0.f, n2 = 0.f, cr = 0.f;
#pragma unroll
    for (int e = 0; e < 8; ++e) {
        x1[e] = r1[base + e];
        x2[e] = r2[base + e];
        n1 += x1[e] * x1[e];
        n2 += x2[e] * x2[e];
        cr += x1[e] * x2[e];
    }
#pragma unroll
    for (int m = 1; m < 32; m <<= 1) {
        n1 += __shfl_xor(n1, m, 32);
        n2 += __shfl_xor(n2, m, 32);
        cr += __shfl_xor(cr, m, 32);
    }
    const float i1 = 1.0f / fmaxf(sqrtf(n1), EPSN);
    const float i2 = 1.0f / fmaxf(sqrtf(n2), EPSN);

    unsigned short* o1 = zbf + (size_t)wave * DIM;
    unsigned short* o2 = zbf + (size_t)(wave + B) * DIM;
#pragma unroll
    for (int e = 0; e < 8; ++e) {
        o1[base + e] = f2bf_bits(x1[e] * i1);
        o2[base + e] = f2bf_bits(x2[e] * i2);
    }
    if (lane == 0) dpos[wave] = cr * i1 * i2;
}

// ------------------------------------------------------------------
// B-tile loader: ISA B 32x16 bf16 layout — lane = 16h + n holds
// column n, K = 8h..8h+7 and 16+8h..16+8h+7 -> two aligned 16B chunks per K-slab.
// ------------------------------------------------------------------
static __device__ __forceinline__ void loadBtile(const unsigned short* __restrict__ zbf,
                                                 int colBase, int ln, int h,
                                                 v16bf B[8]) {
    const unsigned short* Bcol = zbf + (size_t)(colBase + ln) * DIM;
#pragma unroll
    for (int kc = 0; kc < 8; ++kc) {
        v8bf lo = *(const v8bf*)(Bcol + kc * 32 + 8 * h);
        v8bf hi = *(const v8bf*)(Bcol + kc * 32 + 16 + 8 * h);
        B[kc] = __builtin_shufflevector(lo, hi, 0, 1, 2, 3, 4, 5, 6, 7,
                                                8, 9, 10, 11, 12, 13, 14, 15);
    }
}

// 16x16 tile: 8 WMMAs over K=256 with two independent accumulator
// chains, fused exp(10*s), scalar-uniform diagonal mask, row partials.
static __device__ __forceinline__ void tileAccum(const v16bf A[8], const v16bf B[8],
                                                 int rowBase /*SGPR*/, int colBase /*SGPR*/,
                                                 int ln, int h, float rs[8]) {
    v8f acc0 = {0.f, 0.f, 0.f, 0.f, 0.f, 0.f, 0.f, 0.f};
    v8f acc1 = {0.f, 0.f, 0.f, 0.f, 0.f, 0.f, 0.f, 0.f};
#pragma unroll
    for (int kc = 0; kc < 8; kc += 2) {
        acc0 = __builtin_amdgcn_wmma_f32_16x16x32_bf16(
                   false, A[kc],     false, B[kc],     (short)0, acc0, false, false);
        acc1 = __builtin_amdgcn_wmma_f32_16x16x32_bf16(
                   false, A[kc + 1], false, B[kc + 1], (short)0, acc1, false, false);
    }
    if (colBase == rowBase) {              // scalar branch: diagonal tile
#pragma unroll
        for (int r = 0; r < 8; ++r) {
            float v = __expf((acc0[r] + acc1[r]) * INV_T);
            v = (r + 8 * h == ln) ? 0.f : v;   // per-lane cndmask, no branch
            rs[r] += v;
        }
    } else {
#pragma unroll
        for (int r = 0; r < 8; ++r)
            rs[r] += __expf((acc0[r] + acc1[r]) * INV_T);
    }
}

// ------------------------------------------------------------------
// Kernel 2: fused Gram * exp * mask * rowsum.
// Each wave: 16 rows (rowTile) x (N/COL_CHUNKS) columns, K = 256.
// A strip register-cached; B tiles software-pipelined (branch-free
// steady state) so the 16-load clause overlaps the 8-WMMA block with
// partial loadcnt waits.
// ------------------------------------------------------------------
__global__ void __launch_bounds__(256)
infonce_gram(const unsigned short* __restrict__ zbf,
             float* __restrict__ neg_part, int N) {
    // wave id via readfirstlane -> SGPR: rowTile/chunk/colBase uniform,
    // address math on SALU, diagonal test becomes a scalar branch.
    const int wid = __builtin_amdgcn_readfirstlane(
                        (int)((blockIdx.x * blockDim.x + threadIdx.x) >> 5));
    const int lane    = threadIdx.x & 31;
    const int rowTile = wid >> 4;               // wid / COL_CHUNKS
    const int chunk   = wid & (COL_CHUNKS - 1);
    const int rowBase = rowTile * 16;
    const int h       = lane >> 4;              // lane half (K-group select)
    const int ln      = lane & 15;              // row/col within tile

    // ---- preload A strip: row = rowBase + ln, full K = 256 ----
    // ISA 16-bit A 16x32 layout: lane half h holds K = 8h..8h+7 (v0-3)
    // and K = 16+8h..16+8h+7 (v4-7) -> two aligned 16B chunks per slab.
    const unsigned short* Arow = zbf + (size_t)(rowBase + ln) * DIM;
    v16bf A[8];
#pragma unroll
    for (int kc = 0; kc < 8; ++kc) {
        v8bf lo = *(const v8bf*)(Arow + kc * 32 + 8 * h);
        v8bf hi = *(const v8bf*)(Arow + kc * 32 + 16 + 8 * h);
        A[kc] = __builtin_shufflevector(lo, hi, 0, 1, 2, 3, 4, 5, 6, 7,
                                                8, 9, 10, 11, 12, 13, 14, 15);
    }

    const int colsPerChunk = N / COL_CHUNKS;    // 512
    const int colBase0     = chunk * colsPerChunk;
    const int nTiles       = colsPerChunk / 16; // 32 (even, >= 4)

    float rs[8];
#pragma unroll
    for (int r = 0; r < 8; ++r) rs[r] = 0.f;

    // software pipeline, branch-free steady state:
    //   tile(Ba); load Ba'; tile(Bb); load Bb'
    v16bf Ba[8], Bb[8];
    loadBtile(zbf, colBase0, ln, h, Ba);
    loadBtile(zbf, colBase0 + 16, ln, h, Bb);
    int ct = 0;
    for (; ct + 2 < nTiles; ct += 2) {
        tileAccum(A, Ba, rowBase, colBase0 + ct * 16, ln, h, rs);
        loadBtile(zbf, colBase0 + (ct + 2) * 16, ln, h, Ba);
        tileAccum(A, Bb, rowBase, colBase0 + (ct + 1) * 16, ln, h, rs);
        loadBtile(zbf, colBase0 + (ct + 3) * 16, ln, h, Bb);
    }
    tileAccum(A, Ba, rowBase, colBase0 + ct * 16, ln, h, rs);
    tileAccum(A, Bb, rowBase, colBase0 + (ct + 1) * 16, ln, h, rs);

    // reduce over the 16 lanes (n) inside each half; masks 1..8 keep h fixed
#pragma unroll
    for (int r = 0; r < 8; ++r) {
        float v = rs[r];
        v += __shfl_xor(v, 1, 32);
        v += __shfl_xor(v, 2, 32);
        v += __shfl_xor(v, 4, 32);
        v += __shfl_xor(v, 8, 32);
        if (ln == 0)
            neg_part[(size_t)chunk * N + rowBase + r + 8 * h] = v;
    }
}

// ------------------------------------------------------------------
// Kernel 3: loss = mean_i(log negsum_i) - mean_i(log pos_i)
//         = [ sum log(sum_c part[c][i]) - 2 * (1/T) * sum_i d_i ] / N
// ------------------------------------------------------------------
__global__ void infonce_loss(const float* __restrict__ neg_part,
                             const float* __restrict__ dpos,
                             float* __restrict__ out, int B) {
    const int N = 2 * B;
    const int t = threadIdx.x;
    float accL = 0.f, accD = 0.f;
    for (int i = t; i < N; i += blockDim.x) {
        float s = 0.f;
        for (int c = 0; c < COL_CHUNKS; ++c) s += neg_part[(size_t)c * N + i];
        accL += logf(s);
    }
    for (int i = t; i < B; i += blockDim.x) accD += dpos[i];

    __shared__ float sl[256];
    __shared__ float sd[256];
    sl[t] = accL; sd[t] = accD;
    __syncthreads();
    for (int s = 128; s > 0; s >>= 1) {
        if (t < s) { sl[t] += sl[t + s]; sd[t] += sd[t + s]; }
        __syncthreads();
    }
    if (t == 0) out[0] = (sl[0] - 2.0f * INV_T * sd[0]) / (float)N;
}

// ------------------------------------------------------------------
extern "C" void kernel_launch(void* const* d_in, const int* in_sizes, int n_in,
                              void* d_out, int out_size, void* d_ws, size_t ws_size,
                              hipStream_t stream) {
    const float* z1 = (const float*)d_in[0];
    const float* z2 = (const float*)d_in[1];
    const int B = in_sizes[0] / DIM;   // 4096
    const int N = 2 * B;               // 8192

    // workspace layout (all offsets 4KB-multiple aligned):
    unsigned short* zbf = (unsigned short*)d_ws;                         // N*DIM bf16  (4 MB)
    float* neg_part = (float*)((char*)d_ws +
                               (size_t)N * DIM * sizeof(unsigned short)); // 16*N f32   (512 KB)
    float* dpos = (float*)((char*)neg_part +
                           (size_t)COL_CHUNKS * N * sizeof(float));       // B f32      (16 KB)

    const int threads = 256;                       // 8 waves/block (wave32)
    // 1) normalize: B waves
    infonce_normalize<<<dim3((B * 32) / threads), dim3(threads), 0, stream>>>(
        z1, z2, zbf, dpos, B);
    // 2) gram: (N/16)*COL_CHUNKS waves = 8192 waves -> 1024 blocks
    const int totalWaves = (N / 16) * COL_CHUNKS;
    infonce_gram<<<dim3((totalWaves * 32) / threads), dim3(threads), 0, stream>>>(
        zbf, neg_part, N);
    // 3) final reduction
    infonce_loss<<<dim3(1), dim3(256), 0, stream>>>(neg_part, dpos, (float*)d_out, B);
}
